// GRUADecoder_39513699123350
// MI455X (gfx1250) — matrix operand order repaired and checked by reading, not compiled, not run on hardware
//
#include <hip/hip_runtime.h>
#include <hip/hip_bf16.h>
#include <math.h>

// Problem constants
#define HH 1024
#define BB 64
#define SS 1024
#define VV 50257
#define LL 3

typedef __attribute__((ext_vector_type(16))) __bf16    v16bf;
typedef __attribute__((ext_vector_type(8)))  float     v8f;
typedef __attribute__((ext_vector_type(8)))  unsigned  v8u;

// Pack two fp32 -> two bf16 (truncation) in one dword.
// v_perm_b32: S0 -> bytes 7:4, S1 -> bytes 3:0 of concatenated source;
// sel 0x07060302 picks {hi[31:16], lo[31:16]} in ONE instruction.
__device__ __forceinline__ unsigned pkbf(float lo, float hi) {
#if __has_builtin(__builtin_amdgcn_perm)
    return __builtin_amdgcn_perm(__builtin_bit_cast(unsigned, hi),
                                 __builtin_bit_cast(unsigned, lo),
                                 0x07060302u);
#else
    unsigned a = __builtin_bit_cast(unsigned, lo) >> 16;
    unsigned b = __builtin_bit_cast(unsigned, hi) & 0xFFFF0000u;
    return a | b;
#endif
}

// 16 fp32 (4 x float4, already in registers) -> v16bf fragment (8 x v_perm_b32)
__device__ __forceinline__ v16bf cvt16(float4 q0, float4 q1, float4 q2, float4 q3) {
    v8u u;
    u[0] = pkbf(q0.x, q0.y); u[1] = pkbf(q0.z, q0.w);
    u[2] = pkbf(q1.x, q1.y); u[3] = pkbf(q1.z, q1.w);
    u[4] = pkbf(q2.x, q2.y); u[5] = pkbf(q2.z, q2.w);
    u[6] = pkbf(q3.x, q3.y); u[7] = pkbf(q3.z, q3.w);
    return __builtin_bit_cast(v16bf, u);
}

__device__ __forceinline__ float4 ld4(const float* p) { return *(const float4*)p; }

// ---------------------------------------------------------------------------
// Generic GEMM: C[M,N] = act(A[M,K] @ W[N,K]^T + bias), bf16 WMMA, fp32 acc.
// Wave tile: 16(M) x 64(N). Block = 128 threads (4 waves). grid.x covers N in
// steps of 256 cols, grid.y covers M in steps of 16. Requires K%32==0, M%16==0.
//
// Fragment layouts (CDNA5 ISA 7.12.2, wave32):
//  A 16x32 bf16 : lanes 0-15 row M=lane,   K = {k0..k0+7} U {k0+16..k0+23}
//                 lanes16-31 row M=lane-16,K = {k0+8..k0+15} U {k0+24..k0+31}
//  B 32x16 bf16 : lanes 0-15 col N=lane,   K = {k0..k0+7} U {k0+16..k0+23}
//                 lanes16-31 col N=lane-16,K = {k0+8..k0+15} U {k0+24..k0+31}
//  C/D 16x16 f32: acc[j] -> row (lane>>4)*8 + j, col lane&15
template<int ACT>  // 0 = identity, 1 = tanh
__global__ __launch_bounds__(128) void gemm_wmma_kernel(
    const float* __restrict__ A, const float* __restrict__ W,
    const float* __restrict__ bias, float* __restrict__ C,
    int M, int N, int K)
{
    const int lane = threadIdx.x & 31;
    const int wave = threadIdx.x >> 5;
    const int m0 = blockIdx.y * 16;
    const int n0 = (blockIdx.x * 4 + wave) * 64;
    if (n0 >= N) return;                      // wave-uniform exit (EXEC stays all-1)

    // Hoisted per-lane row pointers
    const float* pa = A + (size_t)(m0 + (lane & 15)) * K + ((lane >> 4) << 3);
    const float* pb[4];
#pragma unroll
    for (int t = 0; t < 4; ++t) {
        int row = n0 + t * 16 + (lane & 15);
        if (row >= N) row = N - 1;            // clamp; clamped cols never stored
        pb[t] = W + (size_t)row * K + ((lane >> 4) << 3);
    }

    v8f acc[4] = {};

#pragma unroll 2
    for (int k0 = 0; k0 < K; k0 += 32) {
        if (k0 + 32 < K) __builtin_prefetch(pb[0] + k0 + 32, 0, 1); // global_prefetch_b8

        // Issue ALL raw loads for this k-step first (keeps loads in flight)
        const float* ap = pa + k0;
        float4 ra0 = ld4(ap +  0), ra1 = ld4(ap +  4);
        float4 ra2 = ld4(ap + 16), ra3 = ld4(ap + 20);
        float4 rb[4][4];
#pragma unroll
        for (int t = 0; t < 4; ++t) {
            const float* bp = pb[t] + k0;
            rb[t][0] = ld4(bp + 0);  rb[t][1] = ld4(bp + 4);
            rb[t][2] = ld4(bp + 16); rb[t][3] = ld4(bp + 20);
        }

        // Convert + matrix multiply
        v16bf afr = cvt16(ra0, ra1, ra2, ra3);
#pragma unroll
        for (int t = 0; t < 4; ++t) {
            v16bf bfr = cvt16(rb[t][0], rb[t][1], rb[t][2], rb[t][3]);
            acc[t] = __builtin_amdgcn_wmma_f32_16x16x32_bf16(
                false, afr, false, bfr, (short)0, acc[t], false, false);
        }
    }

    // Epilogue: bias + activation + store
    const int nc = lane & 15;
    const int mb = (lane >> 4) * 8;
#pragma unroll
    for (int t = 0; t < 4; ++t) {
        const int col = n0 + t * 16 + nc;
        if (col < N) {
            const float bv = bias ? bias[col] : 0.0f;
#pragma unroll
            for (int j = 0; j < 8; ++j) {
                float v = acc[t][j] + bv;
                if (ACT == 1) v = tanhf(v);
                C[(size_t)(m0 + mb + j) * N + col] = v;
            }
        }
    }
}

// ---------------------------------------------------------------------------
__global__ void embed_kernel(const int* __restrict__ seq,
                             const float* __restrict__ emb,
                             float* __restrict__ x) {
    int idx = blockIdx.x * blockDim.x + threadIdx.x;   // B*H
    int b = idx >> 10, h = idx & (HH - 1);
    x[idx] = emb[(size_t)seq[b] * HH + h];
}

__global__ void gru_gates_kernel(const float* __restrict__ gi,
                                 const float* __restrict__ gh,
                                 const float* __restrict__ hprev,
                                 float* __restrict__ hout,
                                 float* __restrict__ xout) {
    int idx = blockIdx.x * blockDim.x + threadIdx.x;   // B*H
    int b = idx >> 10, h = idx & (HH - 1);
    const float* gib = gi + (size_t)b * 3 * HH;
    const float* ghb = gh + (size_t)b * 3 * HH;
    float ir = gib[h], iz = gib[HH + h], inn = gib[2 * HH + h];
    float hr = ghb[h], hz = ghb[HH + h], hnn = ghb[2 * HH + h];
    float r = 1.0f / (1.0f + expf(-(ir + hr)));
    float z = 1.0f / (1.0f + expf(-(iz + hz)));
    float n = tanhf(inn + r * hnn);
    float v = (1.0f - z) * n + z * hprev[idx];
    hout[idx] = v;
    xout[idx] = v;
}

// scores[b,s] = dot(rnn[b,:], enc[s,b,:]) ; one wave per (b,s), float4 loads
__global__ void scores_kernel(const float* __restrict__ rnn,
                              const float* __restrict__ enc,
                              float* __restrict__ scores) {
    int wid  = (blockIdx.x * blockDim.x + threadIdx.x) >> 5;
    int lane = threadIdx.x & 31;
    int b = wid >> 10, s = wid & (SS - 1);
    const float4* e4 = (const float4*)(enc + ((size_t)s * BB + b) * HH);
    const float4* r4 = (const float4*)(rnn + (size_t)b * HH);
    float acc = 0.0f;
#pragma unroll
    for (int i = lane; i < HH / 4; i += 32) {
        float4 ev = e4[i], rv = r4[i];
        acc += ev.x * rv.x + ev.y * rv.y + ev.z * rv.z + ev.w * rv.w;
    }
#pragma unroll
    for (int off = 16; off; off >>= 1) acc += __shfl_xor(acc, off, 32);
    if (lane == 0) scores[(size_t)b * SS + s] = acc;
}

__global__ __launch_bounds__(256) void softmax_kernel(const float* __restrict__ scores,
                                                      float* __restrict__ attn) {
    __shared__ float red[8];
    const int b = blockIdx.x, tid = threadIdx.x;
    const int lane = tid & 31, wave = tid >> 5;
    float v[4];
    float m = -3.4e38f;
#pragma unroll
    for (int i = 0; i < 4; ++i) {
        v[i] = scores[(size_t)b * SS + i * 256 + tid];
        m = fmaxf(m, v[i]);
    }
#pragma unroll
    for (int off = 16; off; off >>= 1) m = fmaxf(m, __shfl_xor(m, off, 32));
    if (lane == 0) red[wave] = m;
    __syncthreads();
    m = red[0];
#pragma unroll
    for (int i = 1; i < 8; ++i) m = fmaxf(m, red[i]);
    __syncthreads();
    float s = 0.0f;
#pragma unroll
    for (int i = 0; i < 4; ++i) { v[i] = expf(v[i] - m); s += v[i]; }
#pragma unroll
    for (int off = 16; off; off >>= 1) s += __shfl_xor(s, off, 32);
    if (lane == 0) red[wave] = s;
    __syncthreads();
    float tot = 0.0f;
#pragma unroll
    for (int i = 0; i < 8; ++i) tot += red[i];
    const float inv = 1.0f / tot;
#pragma unroll
    for (int i = 0; i < 4; ++i)
        attn[(size_t)b * SS + i * 256 + tid] = v[i] * inv;
}

__global__ void zero_kernel(float* __restrict__ p) {
    p[blockIdx.x * blockDim.x + threadIdx.x] = 0.0f;
}

// context[b, 4t..4t+3] += sum_{s in chunk} attn[b,s]*enc[s,b,4t..4t+3]
// grid = (B, S/128), block = 256 ; float4 per thread
__global__ __launch_bounds__(256) void context_kernel(const float* __restrict__ attn,
                                                      const float* __restrict__ enc,
                                                      float* __restrict__ ctx) {
    const int b = blockIdx.x;
    const int s0 = blockIdx.y * 128;
    const int t = threadIdx.x;
    float4 acc = {0.f, 0.f, 0.f, 0.f};
    for (int s = s0; s < s0 + 128; ++s) {
        const float w = attn[(size_t)b * SS + s];
        const float4 e = *(const float4*)(enc + ((size_t)s * BB + b) * HH + t * 4);
        acc.x += w * e.x; acc.y += w * e.y; acc.z += w * e.z; acc.w += w * e.w;
    }
    float* c = ctx + (size_t)b * HH + t * 4;
    atomicAdd(c + 0, acc.x);
    atomicAdd(c + 1, acc.y);
    atomicAdd(c + 2, acc.z);
    atomicAdd(c + 3, acc.w);
}

__global__ void concat_kernel(const float* __restrict__ rnn,
                              const float* __restrict__ ctx,
                              float* __restrict__ cin) {
    int idx = blockIdx.x * blockDim.x + threadIdx.x;   // B * 2H
    int b = idx >> 11, c = idx & (2 * HH - 1);
    cin[idx] = (c < HH) ? rnn[(size_t)b * HH + c] : ctx[(size_t)b * HH + (c - HH)];
}

// ---------------------------------------------------------------------------
extern "C" void kernel_launch(void* const* d_in, const int* in_sizes, int n_in,
                              void* d_out, int out_size, void* d_ws, size_t ws_size,
                              hipStream_t stream) {
    const int*   input_seq = (const int*)  d_in[0];
    const float* last_hid  = (const float*)d_in[1];   // (L,B,H)
    const float* enc       = (const float*)d_in[2];   // (S,B,H)
    const float* emb       = (const float*)d_in[3];   // (V,H)
    const float* w_ih      = (const float*)d_in[4];   // (L,3H,H)
    const float* w_hh      = (const float*)d_in[5];   // (L,3H,H)
    const float* b_ih      = (const float*)d_in[6];   // (L,3H)
    const float* b_hh      = (const float*)d_in[7];   // (L,3H)
    const float* concat_w  = (const float*)d_in[8];   // (H,2H)
    const float* concat_b  = (const float*)d_in[9];   // (H)
    const float* out_w     = (const float*)d_in[10];  // (V,H)
    const float* out_b     = (const float*)d_in[11];  // (V)

    float* out_logits = (float*)d_out;                              // (B,V)
    float* out_hidden = out_logits + (size_t)BB * VV;               // (L,B,H)
    float* out_attn   = out_hidden + (size_t)LL * BB * HH;          // (B,1,S)

    // Workspace layout (floats)
    float* ws   = (float*)d_ws;
    float* x    = ws;                        // B*H      (layer input / rnn_output)
    float* gi   = x    + BB * HH;            // B*3H
    float* gh   = gi   + BB * 3 * HH;        // B*3H
    float* scr  = gh   + BB * 3 * HH;        // B*S
    float* ctx  = scr  + BB * SS;            // B*H
    float* cin  = ctx  + BB * HH;            // B*2H
    float* cout = cin  + BB * 2 * HH;        // B*H

    // 1) embedding gather
    embed_kernel<<<(BB * HH) / 256, 256, 0, stream>>>(input_seq, emb, x);

    // 2) 3 GRU layers (sequential)
    for (int l = 0; l < LL; ++l) {
        const float* hprev = last_hid + (size_t)l * BB * HH;
        dim3 gg((3 * HH + 255) / 256, BB / 16);
        gemm_wmma_kernel<0><<<gg, 128, 0, stream>>>(
            x,     w_ih + (size_t)l * 3 * HH * HH, b_ih + (size_t)l * 3 * HH, gi,
            BB, 3 * HH, HH);
        gemm_wmma_kernel<0><<<gg, 128, 0, stream>>>(
            hprev, w_hh + (size_t)l * 3 * HH * HH, b_hh + (size_t)l * 3 * HH, gh,
            BB, 3 * HH, HH);
        gru_gates_kernel<<<(BB * HH) / 256, 256, 0, stream>>>(
            gi, gh, hprev, out_hidden + (size_t)l * BB * HH, x);
    }

    // 3) attention scores + softmax (attn written straight into d_out)
    scores_kernel<<<(BB * SS * 32) / 256, 256, 0, stream>>>(x, enc, scr);
    softmax_kernel<<<BB, 256, 0, stream>>>(scr, out_attn);

    // 4) context = attn @ enc (atomic partial sums over S chunks)
    zero_kernel<<<(BB * HH) / 256, 256, 0, stream>>>(ctx);
    context_kernel<<<dim3(BB, SS / 128), 256, 0, stream>>>(out_attn, enc, ctx);

    // 5) concat + tanh projection
    concat_kernel<<<(BB * 2 * HH) / 256, 256, 0, stream>>>(x, ctx, cin);
    gemm_wmma_kernel<1><<<dim3((HH + 255) / 256, BB / 16), 128, 0, stream>>>(
        cin, concat_w, concat_b, cout, BB, HH, 2 * HH);

    // 6) vocab projection (bandwidth-bound on out_w; fp32 load -> bf16 WMMA)
    gemm_wmma_kernel<0><<<dim3((VV + 255) / 256, BB / 16), 128, 0, stream>>>(
        cout, out_w, out_b, out_logits, BB, VV, HH);
}
